// Gate_16226386444689
// MI455X (gfx1250) — matrix operand order repaired and checked by reading, not compiled, not run on hardware
//
#include <hip/hip_runtime.h>
#include <math.h>

#define T_TOK 16384
#define D_DIM 7168
#define E_EXP 256
#define NEG_INF (-__builtin_inff())

typedef __attribute__((ext_vector_type(16))) __bf16 bf16x16;
typedef __attribute__((ext_vector_type(4)))  __bf16 bf16x4;
typedef __attribute__((ext_vector_type(8)))  float  f32x8;

// Staging buffers (double-buffered bf16 tiles) and the score buffer alias the
// same LDS: staging is dead once the K loop finishes.
struct StageMem {
    __bf16 A[2][128][72];   // 36 KB   (+8 bf16 row pad -> 144B rows, 16B aligned)
    __bf16 B[2][256][72];   // 72 KB
};
union alignas(16) SharedMem {
    StageMem st;            // 108 KB
    float    S[128][260];   // 130 KB  (260 % 8 != 0 -> conflict-free C spill)
};

struct QPair { uint4 lo, hi; };  // 32 B, no padding

// Two aligned 16B LDS loads -> one WMMA bf16 operand (register-level bit_cast,
// keeps the loads as ds_load_b128).
static __device__ __forceinline__ bf16x16 make_frag(const __bf16* p0, const __bf16* p1) {
    QPair q;
    q.lo = *(const uint4*)__builtin_assume_aligned(p0, 16);
    q.hi = *(const uint4*)__builtin_assume_aligned(p1, 16);
    return __builtin_bit_cast(bf16x16, q);
}

// 4x f32 -> bf16, stored as one 8B LDS store (ds_store_b64).
static __device__ __forceinline__ void store_bf4(__bf16* p, float4 v) {
    bf16x4 b;
    b[0] = (__bf16)v.x; b[1] = (__bf16)v.y; b[2] = (__bf16)v.z; b[3] = (__bf16)v.w;
    *(uint2*)__builtin_assume_aligned(p, 8) = __builtin_bit_cast(uint2, b);
}

// ---------------------------------------------------------------------------
// Fused gate: scores = sigmoid(x @ W^T) via bf16 WMMA (f32 accumulate),
// then group-limited top-k routing, all in one kernel.
// Block: M=128 rows x N=256 experts (all), K tiles of 64. 512 thr = 16 waves,
// each wave owns a 32x64 sub-tile (2x4 WMMA tiles, 64 acc VGPRs).
// Double-buffered LDS; global loads for tile k+1 overlap WMMA on tile k.
// ---------------------------------------------------------------------------
__global__ __launch_bounds__(512) void fused_gate_kernel(
    const float* __restrict__ x,      // [T, D]
    const float* __restrict__ W,      // [E, D]
    const float* __restrict__ bias,   // [E]
    float* __restrict__ wout,         // [T, 8]
    int*   __restrict__ iout)         // [T, 8]
{
    __shared__ SharedMem sm;

    const int tid  = threadIdx.x;
    const int lane = tid & 31;
    const int wave = tid >> 5;
    const int wm   = wave & 3;   // 4 M strips of 32 rows
    const int wn   = wave >> 2;  // 4 N strips of 64 experts
    const int rowBase = blockIdx.x * 128;

    const int r16  = lane & 15;
    const int half = lane >> 4;

    const f32x8 zero = {0.f, 0.f, 0.f, 0.f, 0.f, 0.f, 0.f, 0.f};
    f32x8 acc[2][4];
#pragma unroll
    for (int m = 0; m < 2; ++m)
#pragma unroll
        for (int n = 0; n < 4; ++n) acc[m][n] = zero;

    // staging registers for the next K tile
    float4 ra[4];  // A: 128x64 f32 / 512 thr = 4 float4 each
    float4 rb[8];  // B: 256x64 f32 / 512 thr = 8 float4 each

    auto loadRegs = [&](int kb) {
#pragma unroll
        for (int i = 0; i < 4; ++i) {
            const int t4 = tid + i * 512;
            const float* p = x + (size_t)(rowBase + (t4 >> 4)) * D_DIM + kb + (t4 & 15) * 4;
            ra[i] = *(const float4*)p;
            __builtin_prefetch(p + 128, 0, 1);   // 2 tiles ahead (speculative)
        }
#pragma unroll
        for (int i = 0; i < 8; ++i) {
            const int t4 = tid + i * 512;
            const float* p = W + (size_t)(t4 >> 4) * D_DIM + kb + (t4 & 15) * 4;
            rb[i] = *(const float4*)p;
            __builtin_prefetch(p + 128, 0, 1);
        }
    };

    auto storeLds = [&](int buf) {
#pragma unroll
        for (int i = 0; i < 4; ++i) {
            const int t4 = tid + i * 512;
            store_bf4(&sm.st.A[buf][t4 >> 4][(t4 & 15) * 4], ra[i]);
        }
#pragma unroll
        for (int i = 0; i < 8; ++i) {
            const int t4 = tid + i * 512;
            store_bf4(&sm.st.B[buf][t4 >> 4][(t4 & 15) * 4], rb[i]);
        }
    };

    auto compute = [&](int buf) {
#pragma unroll
        for (int ks = 0; ks < 64; ks += 32) {
            // A frags: lane l<16 holds row l, K {0-7,16-23}; l>=16 holds K {8-15,24-31}
            bf16x16 af[2];
#pragma unroll
            for (int m = 0; m < 2; ++m) {
                const int row = wm * 32 + m * 16 + r16;
                af[m] = make_frag(&sm.st.A[buf][row][ks + half * 8],
                                  &sm.st.A[buf][row][ks + 16 + half * 8]);
            }
            // B frags: lane l<16 holds column l, K {0-7,16-23}; l>=16 same column, K {8-15,24-31}
            bf16x16 bfr[4];
#pragma unroll
            for (int n = 0; n < 4; ++n) {
                const int col = wn * 64 + n * 16 + r16;
                bfr[n] = make_frag(&sm.st.B[buf][col][ks + half * 8],
                                   &sm.st.B[buf][col][ks + 16 + half * 8]);
            }
#pragma unroll
            for (int m = 0; m < 2; ++m)
#pragma unroll
                for (int n = 0; n < 4; ++n)
                    acc[m][n] = __builtin_amdgcn_wmma_f32_16x16x32_bf16(
                        /*neg_a=*/false, af[m], /*neg_b=*/false, bfr[n],
                        /*c_mod=*/(short)0, acc[m][n],
                        /*reuse_a=*/false, /*reuse_b=*/false);
        }
    };

    // ---- GEMM main loop (double-buffered) ----
    loadRegs(0);
    storeLds(0);
    __syncthreads();

    int buf = 0;
    for (int kb = 64; kb < D_DIM; kb += 64) {
        loadRegs(kb);        // global loads for tile k+1 (latency hidden below)
        compute(buf);        // WMMA on tile k
        storeLds(buf ^ 1);   // cvt + stage tile k+1 into the other buffer
        __syncthreads();
        buf ^= 1;
    }
    compute(buf);            // tail tile

    // ---- spill sigmoid scores to LDS (aliases dead staging buffers) ----
    __syncthreads();   // all waves done reading staging LDS
#pragma unroll
    for (int m = 0; m < 2; ++m)
#pragma unroll
        for (int n = 0; n < 4; ++n)
#pragma unroll
            for (int i = 0; i < 8; ++i) {
                // C layout: VGPR i -> M = i + 8*(lane>=16), N = lane&15
                const int lr = wm * 32 + m * 16 + i + 8 * half;
                const int lc = wn * 64 + n * 16 + r16;
                const float s = acc[m][n][i];
                sm.S[lr][lc] = 1.0f / (1.0f + __expf(-s));
            }
    __syncthreads();

    // ---- fused routing: each wave handles 8 rows, one row at a time ----
    const int ebase = lane * 8;  // 8 consecutive experts, all in group lane>>2
    float bl[8];
    {
        const float4 b0 = *(const float4*)(bias + ebase);
        const float4 b1 = *(const float4*)(bias + ebase + 4);
        bl[0] = b0.x; bl[1] = b0.y; bl[2] = b0.z; bl[3] = b0.w;
        bl[4] = b1.x; bl[5] = b1.y; bl[6] = b1.z; bl[7] = b1.w;
    }

#pragma unroll 1
    for (int rr = 0; rr < 8; ++rr) {
        const int lr   = wave * 8 + rr;
        const int grow = rowBase + lr;

        float orig[8], sb[8];
        {
            const float4 s0 = *(const float4*)__builtin_assume_aligned(&sm.S[lr][ebase], 16);
            const float4 s1 = *(const float4*)__builtin_assume_aligned(&sm.S[lr][ebase + 4], 16);
            orig[0] = s0.x; orig[1] = s0.y; orig[2] = s0.z; orig[3] = s0.w;
            orig[4] = s1.x; orig[5] = s1.y; orig[6] = s1.z; orig[7] = s1.w;
        }
#pragma unroll
        for (int j = 0; j < 8; ++j) sb[j] = orig[j] + bl[j];

        // per-lane top-2 of its 8 (biased) scores
        float m1 = NEG_INF, m2 = NEG_INF;
#pragma unroll
        for (int j = 0; j < 8; ++j) {
            const float v = sb[j];
            if (v > m1) { m2 = m1; m1 = v; }
            else if (v > m2) { m2 = v; }
        }
        // merge top-2 across the 4 lanes covering one 32-expert group
#pragma unroll
        for (int off = 1; off < 4; off <<= 1) {
            const float o1 = __shfl_xor(m1, off);
            const float o2 = __shfl_xor(m2, off);
            const float hi = fmaxf(m1, o1);
            const float lo = fminf(m1, o1);
            m1 = hi;
            m2 = fmaxf(lo, fmaxf(m2, o2));
        }
        const float gsum = m1 + m2;  // group score (sum of top-2)

        // broadcast all 8 group scores, pick top-4 groups (ties -> lowest idx)
        float gs[8];
#pragma unroll
        for (int g = 0; g < 8; ++g) gs[g] = __shfl(gsum, g * 4);
        unsigned keep = 0;
#pragma unroll
        for (int it = 0; it < 4; ++it) {
            float best = NEG_INF; int bi = 0;
#pragma unroll
            for (int g = 0; g < 8; ++g)
                if (gs[g] > best) { best = gs[g]; bi = g; }
            keep |= 1u << bi;
            gs[bi] = NEG_INF;
        }

        // mask non-selected groups
        const bool inkeep = (keep >> (lane >> 2)) & 1u;
        float ms[8];
#pragma unroll
        for (int j = 0; j < 8; ++j) ms[j] = inkeep ? sb[j] : NEG_INF;

        // 8 rounds of wave-wide argmax (ties -> lowest index, as lax.top_k)
        float wk[8]; int ik[8]; float wsum = 0.f;
#pragma unroll
        for (int k = 0; k < 8; ++k) {
            float lv = NEG_INF; int lj = 0;
#pragma unroll
            for (int j = 0; j < 8; ++j)
                if (ms[j] > lv) { lv = ms[j]; lj = j; }
            int li = ebase + lj;
#pragma unroll
            for (int off = 16; off >= 1; off >>= 1) {
                const float ov = __shfl_xor(lv, off);
                const int   oi = __shfl_xor(li, off);
                if (ov > lv || (ov == lv && oi < li)) { lv = ov; li = oi; }
            }
            const int owner = li >> 3;
            const int slot  = li & 7;
#pragma unroll
            for (int j = 0; j < 8; ++j)
                if (lane == owner && j == slot) ms[j] = NEG_INF;  // remove winner
            float cand = 0.f;
#pragma unroll
            for (int j = 0; j < 8; ++j)
                cand = (j == slot) ? orig[j] : cand;   // slot is wave-uniform
            const float w = __shfl(cand, owner);        // ORIGINAL sigmoid score
            wk[k] = w; ik[k] = li; wsum += w;
        }

        const float scale = 2.5f / wsum;  // normalize then * ROUTE_SCALE
#pragma unroll
        for (int k = 0; k < 8; ++k) {
            if (lane == k) {
                wout[(size_t)grow * 8 + k] = wk[k] * scale;
                iout[(size_t)grow * 8 + k] = ik[k];
            }
        }
    }
}

// ---------------------------------------------------------------------------
extern "C" void kernel_launch(void* const* d_in, const int* in_sizes, int n_in,
                              void* d_out, int out_size, void* d_ws, size_t ws_size,
                              hipStream_t stream) {
    const float* x    = (const float*)d_in[0];   // [16384, 7168]
    const float* W    = (const float*)d_in[1];   // [256, 7168]
    const float* bias = (const float*)d_in[2];   // [256]

    float* wout = (float*)d_out;                     // [T, 8] f32
    int*   iout = (int*)d_out + (size_t)T_TOK * 8;   // [T, 8] i32 (bit-stored)

    fused_gate_kernel<<<T_TOK / 128, 512, 0, stream>>>(x, W, bias, wout, iout);
}
